// Transformer_52819507806811
// MI455X (gfx1250) — matrix operand order repaired and checked by reading, not compiled, not run on hardware
//
#include <hip/hip_runtime.h>
#include <stdint.h>

typedef __attribute__((ext_vector_type(16))) __bf16 v16bf;
typedef __attribute__((ext_vector_type(8)))  float  v8f;
typedef __attribute__((ext_vector_type(4)))  float  f4v;

union BFrag { v16bf v; unsigned short u[16]; uint32_t d[8]; };

// hardware float->bf16 convert (RNE) instead of manual bit rounding
__device__ __forceinline__ unsigned short f2bf(float f) {
    union { __bf16 b; unsigned short u; } c;
    c.b = (__bf16)f;
    return c.u;
}

// monotonic float <-> uint mapping for atomicMax on floats
__device__ __forceinline__ unsigned int enc_m(float f) {
    union { float f; uint32_t u; } c; c.f = f;
    return (c.u & 0x80000000u) ? ~c.u : (c.u | 0x80000000u);
}
__device__ __forceinline__ float dec_m(unsigned int u) {
    union { uint32_t u; float f; } c;
    c.u = (u & 0x80000000u) ? (u ^ 0x80000000u) : ~u;
    return c.f;
}

// ---------------------------------------------------------------------------
// Repack a row-major f32 weight [K x 128] into bf16 WMMA B-fragment layout.
// Frag index = ((nt*KT + kt)*32 + lane)*16 + j ; N = nt*16 + (lane&15),
// K = kt*32 + (j>=8 ? 8 : 0) + (lane>=16 ? 8 : 0) + j   (ISA 7.12.2, 16-bit B layout).
// ---------------------------------------------------------------------------
__global__ void conv_weights(const float* __restrict__ W,
                             unsigned short* __restrict__ dst, int KT) {
    int id = blockIdx.x * blockDim.x + threadIdx.x;
    int total = KT * 8 * 32 * 16;
    if (id >= total) return;
    int j    = id & 15;
    int lane = (id >> 4) & 31;
    int ktn  = id >> 9;
    int kt   = ktn % KT;
    int nt   = ktn / KT;
    int n    = nt * 16 + (lane & 15);
    int kk   = kt * 32 + ((j >= 8) ? 8 : 0) + ((lane >= 16) ? 8 : 0) + j;
    dst[id] = f2bf(W[(size_t)kk * 128 + n]);
}

__global__ void init_ms(unsigned int* __restrict__ m, float* __restrict__ s, int n) {
    int i = blockIdx.x * blockDim.x + threadIdx.x;
    if (i < n) { m[i] = 0x007FFFFFu; /* enc(-inf) */ s[i] = 0.0f; }
}

// ---------------------------------------------------------------------------
// q,k,v,skip = x @ {Wq,Wk,Wv,Wskip} + bias.  16 rows/block, wave w owns
// column tile nt=w.  bf16 WMMA, f32 accumulate.
// ---------------------------------------------------------------------------
__global__ void __launch_bounds__(256)
node_gemm(const float* __restrict__ x,
          const unsigned short* __restrict__ fq, const unsigned short* __restrict__ fk,
          const unsigned short* __restrict__ fv, const unsigned short* __restrict__ fs,
          const float* __restrict__ bq, const float* __restrict__ bk,
          const float* __restrict__ bv, const float* __restrict__ bs,
          float* __restrict__ q, float* __restrict__ k, float* __restrict__ v,
          float* __restrict__ outSkip, int nNodes) {
    __shared__ float xs[16 * 128];
    int row0 = blockIdx.x * 16;
    bool fullTile = (row0 + 16 <= nNodes);
    for (int i = threadIdx.x; i < 16 * 128; i += blockDim.x) {
        int r = i >> 7, c = i & 127;
        int rr = row0 + r; if (rr >= nNodes) rr = nNodes - 1;
        xs[i] = x[(size_t)rr * 128 + c];
    }
    __syncthreads();

    int lane  = threadIdx.x & 31;
    int nt    = threadIdx.x >> 5;          // 8 waves -> 8 column tiles
    int mrow  = lane & 15;                 // A layout: M = lane mod 16
    int koffA = (lane >= 16) ? 8 : 0;      // A layout: K half-offset 8

    BFrag A[4];
#pragma unroll
    for (int kt = 0; kt < 4; ++kt)
#pragma unroll
        for (int j = 0; j < 16; ++j) {
            int lk = ((j < 8) ? 0 : 16) + koffA + (j & 7);
            A[kt].u[j] = f2bf(xs[mrow * 128 + kt * 32 + lk]);
        }

    const unsigned short* fr[4] = { fq, fk, fv, fs };
    const float*          br[4] = { bq, bk, bv, bs };
    float*                op[4] = { q, k, v, outSkip };
    int cl    = lane & 15;
    int cbase = nt * 16 + cl;
    int mbase = (lane >= 16) ? 8 : 0;      // C/D layout: lanes 16-31 hold M+8

#pragma unroll
    for (int mtx = 0; mtx < 4; ++mtx) {
        v8f acc = {};
#pragma unroll
        for (int kt = 0; kt < 4; ++kt) {
            BFrag B;
            const uint32_t* sp =
                (const uint32_t*)(fr[mtx] + ((size_t)(nt * 4 + kt) * 32 + lane) * 16);
#pragma unroll
            for (int d = 0; d < 8; ++d) B.d[d] = sp[d];
            acc = __builtin_amdgcn_wmma_f32_16x16x32_bf16(
                      false, A[kt].v, false, B.v, (short)0, acc, false, false);
        }
        float bias = br[mtx][cbase];
        float* o = op[mtx];
        if (fullTile) {                    // block-uniform: no exec-mask churn
#pragma unroll
            for (int r = 0; r < 8; ++r)
                o[(size_t)(row0 + mbase + r) * 128 + cbase] = acc[r] + bias;
        } else {
#pragma unroll
            for (int r = 0; r < 8; ++r) {
                int mr = row0 + mbase + r;
                if (mr < nNodes) o[(size_t)mr * 128 + cbase] = acc[r] + bias;
            }
        }
    }
}

// Build A-frags for one 16-edge tile from edge_attr [E x 64] (f32 -> bf16).
// Non-temporal: edge_attr is a 410MB single-use stream; keep L2 for q/k/v.
__device__ __forceinline__ void load_edge_A(const float* __restrict__ ea,
                                            int e0, int lane, int E, BFrag A[2]) {
    int mrow  = lane & 15;
    int koffA = (lane >= 16) ? 8 : 0;
    int er = e0 + mrow; if (er >= E) er = E - 1;
#pragma unroll
    for (int kt = 0; kt < 2; ++kt) {
        const float* rowp = ea + (size_t)er * 64 + kt * 32 + koffA;
        f4v f0 = __builtin_nontemporal_load((const f4v*)(rowp + 0));
        f4v f1 = __builtin_nontemporal_load((const f4v*)(rowp + 4));
        f4v f2 = __builtin_nontemporal_load((const f4v*)(rowp + 16));
        f4v f3 = __builtin_nontemporal_load((const f4v*)(rowp + 20));
        float f[16];
        *(f4v*)(f + 0) = f0; *(f4v*)(f + 4) = f1;
        *(f4v*)(f + 8) = f2; *(f4v*)(f + 12) = f3;
#pragma unroll
        for (int j = 0; j < 16; ++j) A[kt].u[j] = f2bf(f[j]);
    }
}

// ---------------------------------------------------------------------------
// Pass 1 over edges: e = ea@We (WMMA), alpha = <q[dst], k[src]+e>/8 per head,
// store alpha, atomicMax into m.  One wave per 16-edge tile.
// ---------------------------------------------------------------------------
__global__ void __launch_bounds__(256)
edge_alpha(const float* __restrict__ ea, const long long* __restrict__ esrc,
           const long long* __restrict__ edst, const unsigned short* __restrict__ fwe,
           const float* __restrict__ q, const float* __restrict__ k,
           float* __restrict__ alphaA, unsigned int* __restrict__ m,
           int nTiles, int E) {
    int wave = blockIdx.x * (blockDim.x >> 5) + (threadIdx.x >> 5);
    if (wave >= nTiles) return;            // wave-uniform: EXEC stays all-ones
    int lane = threadIdx.x & 31;
    int e0 = wave * 16;
    if (e0 + 16 < E) __builtin_prefetch(ea + (size_t)(e0 + 16) * 64, 0, 1);

    BFrag A[2];
    load_edge_A(ea, e0, lane, E, A);

    int mbase = (lane >= 16) ? 8 : 0;
    int cl    = lane & 15;
    long long srcs[8], dsts[8];
#pragma unroll
    for (int r = 0; r < 8; ++r) {
        int ee = e0 + mbase + r; if (ee >= E) ee = E - 1;
        srcs[r] = esrc[ee];
        dsts[r] = edst[ee];
    }
    float part[8][2];
#pragma unroll
    for (int r = 0; r < 8; ++r) { part[r][0] = 0.f; part[r][1] = 0.f; }

#pragma unroll
    for (int nt = 0; nt < 8; ++nt) {
        v8f acc = {};
#pragma unroll
        for (int kt = 0; kt < 2; ++kt) {
            BFrag B;
            const uint32_t* sp =
                (const uint32_t*)(fwe + ((size_t)(nt * 2 + kt) * 32 + lane) * 16);
#pragma unroll
            for (int d = 0; d < 8; ++d) B.d[d] = sp[d];
            acc = __builtin_amdgcn_wmma_f32_16x16x32_bf16(
                      false, A[kt].v, false, B.v, (short)0, acc, false, false);
        }
        int c = nt * 16 + cl;
        int h = nt >> 2;                   // channels 0-63 -> head 0, 64-127 -> head 1
#pragma unroll
        for (int r = 0; r < 8; ++r) {
            float qv = q[(size_t)dsts[r] * 128 + c];
            float kv = k[(size_t)srcs[r] * 128 + c] + acc[r];
            part[r][h] += qv * kv;
        }
    }
#pragma unroll
    for (int r = 0; r < 8; ++r) {
        float p0 = part[r][0], p1 = part[r][1];
#pragma unroll
        for (int d = 1; d < 16; d <<= 1) {
            p0 += __shfl_xor(p0, d, 32);
            p1 += __shfl_xor(p1, d, 32);
        }
        int ee = e0 + mbase + r;
        if (cl == 0 && ee < E) {
            float a0 = p0 * 0.125f, a1 = p1 * 0.125f;   // / sqrt(64)
            alphaA[(size_t)ee * 2 + 0] = a0;
            alphaA[(size_t)ee * 2 + 1] = a1;
            atomicMax(&m[(size_t)dsts[r] * 2 + 0], enc_m(a0));
            atomicMax(&m[(size_t)dsts[r] * 2 + 1], enc_m(a1));
        }
    }
}

__global__ void edge_exp(const long long* __restrict__ edst,
                         const unsigned int* __restrict__ m,
                         float* __restrict__ alphaA, float* __restrict__ s, int E) {
    int i = blockIdx.x * blockDim.x + threadIdx.x;
    if (i >= 2 * E) return;
    int e = i >> 1, h = i & 1;
    long long d = edst[e];
    float mm = dec_m(m[(size_t)d * 2 + h]);
    if (!__builtin_isfinite(mm)) mm = 0.0f;
    float a = __expf(alphaA[i] - mm);
    alphaA[i] = a;
    atomicAdd(&s[(size_t)d * 2 + h], a);
}

// ---------------------------------------------------------------------------
// Pass 2 over edges: recompute e via WMMA, msg = (v[src]+e)*a/(s+eps),
// scatter-add into out (which already holds x@Wskip + bskip).
// ---------------------------------------------------------------------------
__global__ void __launch_bounds__(256)
edge_msg(const float* __restrict__ ea, const long long* __restrict__ esrc,
         const long long* __restrict__ edst, const unsigned short* __restrict__ fwe,
         const float* __restrict__ v, const float* __restrict__ alphaA,
         const float* __restrict__ s, float* __restrict__ out,
         int nTiles, int E) {
    int wave = blockIdx.x * (blockDim.x >> 5) + (threadIdx.x >> 5);
    if (wave >= nTiles) return;
    int lane = threadIdx.x & 31;
    int e0 = wave * 16;
    bool full = (e0 + 16 <= E);
    if (e0 + 16 < E) __builtin_prefetch(ea + (size_t)(e0 + 16) * 64, 0, 1);

    BFrag A[2];
    load_edge_A(ea, e0, lane, E, A);

    int mbase = (lane >= 16) ? 8 : 0;
    int cl    = lane & 15;
    long long srcs[8], dsts[8];
    float coef[8][2];
#pragma unroll
    for (int r = 0; r < 8; ++r) {
        int ee = e0 + mbase + r;
        if (ee >= E) ee = E - 1;
        srcs[r] = esrc[ee];
        dsts[r] = edst[ee];
        float s0 = s[(size_t)dsts[r] * 2 + 0] + 1e-16f;
        float s1 = s[(size_t)dsts[r] * 2 + 1] + 1e-16f;
        coef[r][0] = alphaA[(size_t)ee * 2 + 0] / s0;
        coef[r][1] = alphaA[(size_t)ee * 2 + 1] / s1;
    }

#pragma unroll
    for (int nt = 0; nt < 8; ++nt) {
        v8f acc = {};
#pragma unroll
        for (int kt = 0; kt < 2; ++kt) {
            BFrag B;
            const uint32_t* sp =
                (const uint32_t*)(fwe + ((size_t)(nt * 2 + kt) * 32 + lane) * 16);
#pragma unroll
            for (int d = 0; d < 8; ++d) B.d[d] = sp[d];
            acc = __builtin_amdgcn_wmma_f32_16x16x32_bf16(
                      false, A[kt].v, false, B.v, (short)0, acc, false, false);
        }
        int c = nt * 16 + cl;
        int h = nt >> 2;
        if (full) {
#pragma unroll
            for (int r = 0; r < 8; ++r) {
                float msg = (v[(size_t)srcs[r] * 128 + c] + acc[r]) * coef[r][h];
                atomicAdd(&out[(size_t)dsts[r] * 128 + c], msg);
            }
        } else {
#pragma unroll
            for (int r = 0; r < 8; ++r) {
                if (e0 + mbase + r < E) {
                    float msg = (v[(size_t)srcs[r] * 128 + c] + acc[r]) * coef[r][h];
                    atomicAdd(&out[(size_t)dsts[r] * 128 + c], msg);
                }
            }
        }
    }
}

// ---------------------------------------------------------------------------
extern "C" void kernel_launch(void* const* d_in, const int* in_sizes, int n_in,
                              void* d_out, int out_size, void* d_ws, size_t ws_size,
                              hipStream_t stream) {
    const float*     x    = (const float*)d_in[0];
    const long long* eidx = (const long long*)d_in[1];   // int64 [2, E]
    const float*     ea   = (const float*)d_in[2];
    const float*     Wq   = (const float*)d_in[3];
    const float*     bq   = (const float*)d_in[4];
    const float*     Wk   = (const float*)d_in[5];
    const float*     bk   = (const float*)d_in[6];
    const float*     Wv   = (const float*)d_in[7];
    const float*     bv   = (const float*)d_in[8];
    const float*     We   = (const float*)d_in[9];
    const float*     Ws   = (const float*)d_in[10];
    const float*     bs   = (const float*)d_in[11];
    float* out = (float*)d_out;

    int N = in_sizes[0] / 128;
    int E = in_sizes[1] / 2;
    const long long* esrc = eidx;
    const long long* edst = eidx + E;

    // workspace carve (all regions fully re-written each call)
    char* w = (char*)d_ws;
    auto carve = [&](size_t bytes) -> char* {
        char* p = w; w += (bytes + 255) & ~(size_t)255; return p;
    };
    float*          q      = (float*)carve((size_t)N * 128 * 4);
    float*          k      = (float*)carve((size_t)N * 128 * 4);
    float*          v      = (float*)carve((size_t)N * 128 * 4);
    float*          alphaA = (float*)carve((size_t)E * 2 * 4);
    unsigned int*   m      = (unsigned int*)carve((size_t)N * 2 * 4);
    float*          s      = (float*)carve((size_t)N * 2 * 4);
    unsigned short* fq     = (unsigned short*)carve(4 * 8 * 32 * 16 * 2);
    unsigned short* fk     = (unsigned short*)carve(4 * 8 * 32 * 16 * 2);
    unsigned short* fv     = (unsigned short*)carve(4 * 8 * 32 * 16 * 2);
    unsigned short* fs     = (unsigned short*)carve(4 * 8 * 32 * 16 * 2);
    unsigned short* fe     = (unsigned short*)carve(2 * 8 * 32 * 16 * 2);

    conv_weights<<<(16384 + 255) / 256, 256, 0, stream>>>(Wq, fq, 4);
    conv_weights<<<(16384 + 255) / 256, 256, 0, stream>>>(Wk, fk, 4);
    conv_weights<<<(16384 + 255) / 256, 256, 0, stream>>>(Wv, fv, 4);
    conv_weights<<<(16384 + 255) / 256, 256, 0, stream>>>(Ws, fs, 4);
    conv_weights<<<(8192  + 255) / 256, 256, 0, stream>>>(We, fe, 2);
    init_ms<<<(N * 2 + 255) / 256, 256, 0, stream>>>(m, s, N * 2);

    node_gemm<<<(N + 15) / 16, 256, 0, stream>>>(x, fq, fk, fv, fs,
                                                 bq, bk, bv, bs,
                                                 q, k, v, out, N);

    int nTiles = (E + 15) / 16;
    int blocks = (nTiles + 7) / 8;         // 8 waves per 256-thread block
    edge_alpha<<<blocks, 256, 0, stream>>>(ea, esrc, edst, fe, q, k,
                                           alphaA, m, nTiles, E);
    edge_exp<<<(2 * E + 255) / 256, 256, 0, stream>>>(edst, m, alphaA, s, E);
    edge_msg<<<blocks, 256, 0, stream>>>(ea, esrc, edst, fe, v, alphaA, s,
                                         out, nTiles, E);
}
